// MultiHeadAttention_10187662426119
// MI455X (gfx1250) — matrix operand order repaired and checked by reading, not compiled, not run on hardware
//
#include <hip/hip_runtime.h>
#include <hip/hip_bf16.h>

// ---- problem constants (from reference) ----
#define Bc   16
#define NQc  1024
#define NGc  1024
#define Dc   512
#define Hc   8
#define DKc  64
#define DVc  64
#define Ec   512
#define BNc  (Bc * NQc)   // 16384 rows for q and h alike (NQ == NG)

typedef __attribute__((ext_vector_type(16))) __bf16    v16bf;
typedef __attribute__((ext_vector_type(8)))  float     v8f;
typedef __attribute__((ext_vector_type(4)))  unsigned  u32x4;
typedef __attribute__((ext_vector_type(4)))  float     f32x4;

// ---------------- bf16 helpers (native hardware convert) ----------------
__device__ __forceinline__ unsigned short bf16_bits(float f) {
  __bf16 b = (__bf16)f;                       // native v_cvt, RNE
  return __builtin_bit_cast(unsigned short, b);
}
__device__ __forceinline__ unsigned pack_bf16x2(float lo, float hi) {
  return (unsigned)bf16_bits(lo) | ((unsigned)bf16_bits(hi) << 16);
}

// ---------------- WMMA fragment loaders (ISA 7.12.2 layouts, wave32) ----------------
// A: 16x32 (MxK) bf16. lane m = l&15; VGPR v holds K pair
//    k = 2*(v&3) + (l&16 ? 8 : 0) + (v&4 ? 16 : 0)
// => per lane, VGPRs 0-3 = shorts [khi..khi+7], VGPRs 4-7 = [khi+16..khi+23]
__device__ __forceinline__ v16bf a_frag_bf16(const __bf16* base, int ldr, int lane) {
  int m   = lane & 15;
  int khi = (lane & 16) ? 8 : 0;
  const unsigned short* p = (const unsigned short*)base + (size_t)m * ldr + khi;
  v16bf a;
  ((u32x4*)&a)[0] = *(const u32x4*)(p);
  ((u32x4*)&a)[1] = *(const u32x4*)(p + 16);
  return a;
}

// B: 32x16 (KxN) bf16. lane n = l&15; same K order as A:
// => per lane, VGPRs 0-3 = shorts [koff..koff+7], VGPRs 4-7 = [koff+16..koff+23], koff = (l&16 ? 8 : 0)
// element (k,n) = base[n*ldn + k]
__device__ __forceinline__ v16bf b_frag_bf16(const __bf16* base, int ldn, int lane) {
  int n    = lane & 15;
  int koff = (lane & 16) ? 8 : 0;
  const unsigned short* p = (const unsigned short*)base + (size_t)n * ldn + koff;
  v16bf b;
  ((u32x4*)&b)[0] = *(const u32x4*)(p);
  ((u32x4*)&b)[1] = *(const u32x4*)(p + 16);
  return b;
}

__device__ __forceinline__ v8f wmma_bf16(v16bf a, v16bf b, v8f c) {
  return __builtin_amdgcn_wmma_f32_16x16x32_bf16(false, a, false, b, (short)0, c, false, false);
}

// C/D 16x16 f32: VGPR v -> m = v + (lane&16 ? 8 : 0), n = lane&15

// =====================================================================
// Kernel 0a: q/h f32 -> bf16 straight copy (coalesced, packed stores)
// =====================================================================
__global__ void prep_x_kernel(const float* __restrict__ q, const float* __restrict__ h,
                              __bf16* __restrict__ qx, __bf16* __restrict__ hx) {
  const size_t N = (size_t)BNc * Dc;     // 8,388,608 per tensor
  size_t base = ((size_t)blockIdx.x * blockDim.x + threadIdx.x) * 4;
  if (base >= N) return;
  f32x4 t = *(const f32x4*)(q + base);
  f32x4 u = *(const f32x4*)(h + base);
  unsigned* qo = (unsigned*)(qx + base);
  unsigned* ho = (unsigned*)(hx + base);
  qo[0] = pack_bf16x2(t[0], t[1]);
  qo[1] = pack_bf16x2(t[2], t[3]);
  ho[0] = pack_bf16x2(u[0], u[1]);
  ho[1] = pack_bf16x2(u[2], u[3]);
}

// =====================================================================
// Kernel 0b: weight transpose+convert so every B fragment is K-contiguous bf16
//   Wq/Wk/Wv (H,D,DK) f32 -> Wt (H,DK,D) bf16
//   W_out (H*DV=512, E) f32 -> Wot (E, 512) bf16
// =====================================================================
__global__ void prep_w_kernel(const float* __restrict__ Wq, const float* __restrict__ Wk,
                              const float* __restrict__ Wv, const float* __restrict__ Wo,
                              __bf16* __restrict__ Wqt, __bf16* __restrict__ Wkt,
                              __bf16* __restrict__ Wvt, __bf16* __restrict__ Wot) {
  const int WN = Hc * DKc * Dc;          // 262144
  int i = blockIdx.x * blockDim.x + threadIdx.x;
  if (i < 3 * WN) {
    int sel = i >> 18;                   // / WN
    int r   = i & (WN - 1);
    int hh  = r >> 15;                   // / (DKc*Dc)
    int r2  = r & (DKc * Dc - 1);
    int nk  = r2 >> 9;                   // / Dc
    int kd  = r2 & (Dc - 1);
    const float* W = (sel == 0) ? Wq : (sel == 1 ? Wk : Wv);
    __bf16*     Wt = (sel == 0) ? Wqt : (sel == 1 ? Wkt : Wvt);
    Wt[(size_t)hh * DKc * Dc + (size_t)nk * Dc + kd] =
        (__bf16)W[(size_t)hh * Dc * DKc + (size_t)kd * DKc + nk];
  } else {
    int r = i - 3 * WN;
    if (r < Ec * 512) {
      int e = r >> 9;
      int k = r & 511;
      Wot[(size_t)e * 512 + k] = (__bf16)Wo[(size_t)k * Ec + e];
    }
  }
}

// =====================================================================
// Kernel 1: per-head projections  Qb = q*Wq, Kb = h*Wk, Vt = (h*Wv)^T (bf16)
// one wave per 16x64 output stripe: A loaded once per k-step, 4 WMMAs
// waves = 3 * H * (BN/16) = 24576
// =====================================================================
__global__ void proj_kernel(const __bf16* __restrict__ qx, const __bf16* __restrict__ hx,
                            const __bf16* __restrict__ Wqt, const __bf16* __restrict__ Wkt,
                            const __bf16* __restrict__ Wvt,
                            __bf16* __restrict__ Qb, __bf16* __restrict__ Kb,
                            __bf16* __restrict__ Vt) {
  int wid  = (blockIdx.x * blockDim.x + threadIdx.x) >> 5;
  int lane = threadIdx.x & 31;
  int rt = wid & 1023;         wid >>= 10;       // row tile (0..1023)
  int hh = wid & 7;            wid >>= 3;        // head
  int sel = wid;                                 // 0=Q 1=K 2=V
  if (sel > 2) return;

  const __bf16* X  = (sel == 0) ? qx : hx;
  const __bf16* Wt = (sel == 0) ? Wqt : (sel == 1 ? Wkt : Wvt);

  const __bf16* Arow  = X + (size_t)rt * 16 * Dc;
  const __bf16* Wbase = Wt + (size_t)hh * DKc * Dc;   // 64 rows (n, k), K-contig

  v8f acc[4] = {};
#pragma unroll 4
  for (int kk = 0; kk < Dc; kk += 32) {
    if (kk + 32 < Dc)  // pull next A k-slab toward the WGP ahead of use
      __builtin_prefetch(Arow + (size_t)(lane & 15) * Dc + kk + 32, 0, 1);
    v16bf a = a_frag_bf16(Arow + kk, Dc, lane);
#pragma unroll
    for (int ct = 0; ct < 4; ++ct) {
      v16bf b = b_frag_bf16(Wbase + (size_t)ct * 16 * Dc + kk, Dc, lane);
      acc[ct] = wmma_bf16(a, b, acc[ct]);
    }
  }

  int n  = lane & 15;
  int mb = (lane & 16) ? 8 : 0;
  if (sel == 2) {
    // V stored transposed: Vt[h][b][dv][g], g contiguous -> one b128 store per lane/tile
    int bbb = rt >> 6;             // batch
    int g0  = (rt & 63) * 16;      // g base of this tile
#pragma unroll
    for (int ct = 0; ct < 4; ++ct) {
      unsigned short* o = (unsigned short*)Vt +
          (((size_t)hh * Bc + bbb) * DVc + ct * 16 + n) * NGc + g0 + mb;
      u32x4 w;
#pragma unroll
      for (int i = 0; i < 4; ++i) w[i] = pack_bf16x2(acc[ct][2 * i], acc[ct][2 * i + 1]);
      *(u32x4*)o = w;
    }
  } else {
    __bf16* O = (sel == 0) ? Qb : Kb;
    unsigned short* o =
        (unsigned short*)O + ((size_t)hh * BNc + (size_t)rt * 16) * DKc + n;
#pragma unroll
    for (int ct = 0; ct < 4; ++ct)
#pragma unroll
      for (int v = 0; v < 8; ++v)
        o[(size_t)(mb + v) * DKc + ct * 16] = bf16_bits(acc[ct][v]);
  }
}

// =====================================================================
// Kernel 2: fused attention for one (head, batch, 16-row q-tile).
// 8 waves / 256 threads. LDS: compat f32 stripe (64 KB) + attn bf16 stripe (32 KB).
//   A) compat = scale * Q K^T, mask -> -1e30, into LDS   (WMMA, Q frags hoisted)
//   B) row softmax in LDS; attn f32 -> d_out, attn bf16 -> LDS
//   C) heads = attn(bf16, LDS) * Vt, split-K over waves, LDS-reduced  (WMMA)
// =====================================================================
#define ATTN_SMEM_BYTES (16 * NGc * 4 + 16 * NGc * 2 + 256 * 4 + 16 * 4 + 8 * 256 * 4)
__global__ void attn_kernel(const __bf16* __restrict__ Qb, const __bf16* __restrict__ Kb,
                            const __bf16* __restrict__ Vt,
                            const unsigned char* __restrict__ mask, // jnp bool = 1 byte
                            float* __restrict__ attn_out, __bf16* __restrict__ heads) {
  extern __shared__ char smem_raw[];
  float*  compat = (float*)smem_raw;                               // [16][NG] f32
  __bf16* abf    = (__bf16*)(smem_raw + 16 * NGc * 4);             // [16][NG] bf16
  float*  red    = (float*)(smem_raw + 16 * NGc * 6);              // [16][16]
  float*  rowstat = red + 256;                                     // [16]
  float*  hpart   = rowstat + 16;                                  // [8][256]

  int bid = blockIdx.x;
  int qt = bid & 63;  bid >>= 6;                // q tile (0..63)
  int bb = bid & 15;  bid >>= 4;                // batch
  int hh = bid;                                 // head

  int wave = threadIdx.x >> 5;
  int lane = threadIdx.x & 31;
  int n  = lane & 15;
  int mb = (lane & 16) ? 8 : 0;
  const float scale = 0.125f;                   // 1/sqrt(64)

  const __bf16* Qrow = Qb + ((size_t)hh * BNc + (size_t)bb * NQc + (size_t)qt * 16) * DKc;
  const unsigned char* mrow =
      mask + ((size_t)bb * NQc + (size_t)qt * 16) * NGc;

  // ---- Phase A: compat tiles via WMMA (Q fragments loop-invariant) ----
  v16bf qa0 = a_frag_bf16(Qrow, DKc, lane);
  v16bf qa1 = a_frag_bf16(Qrow + 32, DKc, lane);
  for (int t = wave; t < NGc / 16; t += 8) {
    int g0 = t * 16;
    const __bf16* Krow = Kb + ((size_t)hh * BNc + (size_t)bb * NGc + g0) * DKc;
    if (t + 8 < NGc / 16)   // prefetch next K tile for this wave
      __builtin_prefetch(Krow + (size_t)128 * DKc + (size_t)(lane & 15) * DKc, 0, 1);
    v8f acc = {};
    acc = wmma_bf16(qa0, b_frag_bf16(Krow, DKc, lane), acc);
    acc = wmma_bf16(qa1, b_frag_bf16(Krow + 32, DKc, lane), acc);
#pragma unroll
    for (int v = 0; v < 8; ++v) {
      int m = mb + v;
      float val = acc[v] * scale;
      if (mrow[(size_t)m * NGc + g0 + n]) val = -1e30f;   // exp underflows to exact 0
      compat[m * NGc + g0 + n] = val;
    }
  }
  __syncthreads();

  // ---- Phase B: softmax over NG; attn f32 -> global, attn bf16 -> LDS ----
  int rr = threadIdx.x >> 4;    // row 0..15
  int sg = threadIdx.x & 15;    // 64-wide segment
  float* seg = compat + rr * NGc + sg * 64;

  float mx = -1e30f;
#pragma unroll 8
  for (int i = 0; i < 64; ++i) mx = fmaxf(mx, seg[i]);
  red[rr * 16 + sg] = mx;
  __syncthreads();
  if (sg == 0) {
    float m = red[rr * 16];
#pragma unroll
    for (int i = 1; i < 16; ++i) m = fmaxf(m, red[rr * 16 + i]);
    rowstat[rr] = m;
  }
  __syncthreads();
  float rmax = rowstat[rr];
  float s = 0.f;
#pragma unroll 8
  for (int i = 0; i < 64; ++i) {
    float e = __expf(seg[i] - rmax);
    seg[i] = e;
    s += e;
  }
  __syncthreads();            // everyone done reading rowstat (max) before reuse
  red[rr * 16 + sg] = s;
  __syncthreads();
  if (sg == 0) {
    float t = 0.f;
#pragma unroll
    for (int i = 0; i < 16; ++i) t += red[rr * 16 + i];
    rowstat[rr] = t;
  }
  __syncthreads();
  float inv = 1.0f / fmaxf(rowstat[rr], 1e-37f);
  float* arow = attn_out + (((size_t)hh * Bc + bb) * NQc + (size_t)qt * 16 + rr) * NGc + sg * 64;
  unsigned* ab32 = (unsigned*)(abf + rr * NGc + sg * 64);
#pragma unroll 4
  for (int i = 0; i < 64; i += 4) {
    f32x4 p;
    p[0] = seg[i] * inv;  p[1] = seg[i + 1] * inv;
    p[2] = seg[i + 2] * inv;  p[3] = seg[i + 3] * inv;
    *(f32x4*)(arow + i) = p;            // sole HBM touch of the 537 MB attn tensor
    ab32[(i >> 1)]     = pack_bf16x2(p[0], p[1]);
    ab32[(i >> 1) + 1] = pack_bf16x2(p[2], p[3]);
  }
  __syncthreads();

  // ---- Phase C: heads = attn * V (split-K: wave = dv-tile x half of NG) ----
  int tn   = wave & 3;        // dv tile 0..3
  int half = wave >> 2;       // K half
  const __bf16* Vbase =
      Vt + (((size_t)hh * Bc + bb) * DVc + tn * 16) * NGc;       // (dv, g) rows, K-contig
  v8f acc = {};
  for (int gg = half * (NGc / 2); gg < (half + 1) * (NGc / 2); gg += 32) {
    v16bf a = a_frag_bf16(abf + gg, NGc, lane);                  // LDS bf16, 2x ds_b128
    v16bf b = b_frag_bf16(Vbase + gg, NGc, lane);
    acc = wmma_bf16(a, b, acc);
  }
  float* hp = hpart + wave * 256;
#pragma unroll
  for (int v = 0; v < 8; ++v) hp[(mb + v) * 16 + n] = acc[v];
  __syncthreads();

  if (wave < 4) {
    float* h0 = hpart + wave * 256;
    float* h1 = hpart + (wave + 4) * 256;
    // heads layout: (B*NQ, H*DV) so out-proj A matrix has contiguous K
    unsigned short* ho = (unsigned short*)heads +
        ((size_t)bb * NQc + (size_t)qt * 16) * (Hc * DVc) + hh * DVc + wave * 16 + n;
#pragma unroll
    for (int v = 0; v < 8; ++v) {
      int m = mb + v;
      ho[(size_t)m * (Hc * DVc)] = bf16_bits(h0[m * 16 + n] + h1[m * 16 + n]);
    }
  }
}

// =====================================================================
// Kernel 3: out = heads (B*NQ x 512 bf16) * Wot^T  (Wot is (E,512) bf16, K-contig)
// one wave per 16x64 output stripe (4 E tiles): A loaded once per k-step
// waves = (BN/16) * (E/64) = 1024 * 8 = 8192
// =====================================================================
__global__ void out_kernel(const __bf16* __restrict__ heads,
                           const __bf16* __restrict__ Wot,
                           float* __restrict__ out) {
  int wid  = (blockIdx.x * blockDim.x + threadIdx.x) >> 5;
  int lane = threadIdx.x & 31;
  int cg = wid & 7;   wid >>= 3;   // E group of 4 tiles (0..7)
  int rt = wid;                    // row tile (0..1023)
  if (rt >= BNc / 16) return;

  const __bf16* Arow  = heads + (size_t)rt * 16 * (Hc * DVc);
  const __bf16* Wbase = Wot + (size_t)cg * 64 * 512;   // 64 rows (e, k), K-contig

  v8f acc[4] = {};
#pragma unroll 4
  for (int kk = 0; kk < Hc * DVc; kk += 32) {
    if (kk + 32 < Hc * DVc)
      __builtin_prefetch(Arow + (size_t)(lane & 15) * (Hc * DVc) + kk + 32, 0, 1);
    v16bf a = a_frag_bf16(Arow + kk, Hc * DVc, lane);
#pragma unroll
    for (int j = 0; j < 4; ++j) {
      v16bf b = b_frag_bf16(Wbase + (size_t)j * 16 * 512 + kk, 512, lane);
      acc[j] = wmma_bf16(a, b, acc[j]);
    }
  }

  int n  = lane & 15;
  int mb = (lane & 16) ? 8 : 0;
  float* o = out + ((size_t)rt * 16) * Ec + cg * 64 + n;
#pragma unroll
  for (int j = 0; j < 4; ++j)
#pragma unroll
    for (int v = 0; v < 8; ++v) o[(size_t)(mb + v) * Ec + j * 16] = acc[j][v];
}

// =====================================================================
extern "C" void kernel_launch(void* const* d_in, const int* in_sizes, int n_in,
                              void* d_out, int out_size, void* d_ws, size_t ws_size,
                              hipStream_t stream) {
  const float* q   = (const float*)d_in[0];
  const float* h   = (const float*)d_in[1];
  const float* Wq  = (const float*)d_in[2];
  const float* Wk  = (const float*)d_in[3];
  const float* Wv  = (const float*)d_in[4];
  const float* Wo  = (const float*)d_in[5];
  const unsigned char* mask = (const unsigned char*)d_in[6]; // jnp bool = 1 byte/elem

  // d_out = [ out (B*NQ*E f32) | attn (H*B*NQ*NG f32) ]
  float* out_p  = (float*)d_out;
  float* attn_p = out_p + (size_t)Bc * NQc * Ec;

  // workspace (all bf16):
  //  qx, hx         : BN*D each            (16 MB ea)
  //  Qb, Kb         : H*BN*DK each         (16 MB ea)
  //  Vt             : H*B*DV*NG            (16 MB)
  //  heads          : BN*(H*DV)            (16 MB)
  //  Wqt,Wkt,Wvt    : H*DK*D each          (0.5 MB ea)
  //  Wot            : E*512                (0.5 MB)      total ~98 MB
  const size_t big = (size_t)8 * 1024 * 1024;            // 8,388,608 elements
  __bf16* qx    = (__bf16*)d_ws;
  __bf16* hx    = qx + big;
  __bf16* Qb    = hx + big;
  __bf16* Kb    = Qb + big;
  __bf16* Vt    = Kb + big;
  __bf16* heads = Vt + big;
  __bf16* Wqt   = heads + big;
  __bf16* Wkt   = Wqt + (size_t)Hc * DKc * Dc;
  __bf16* Wvt   = Wkt + (size_t)Hc * DKc * Dc;
  __bf16* Wot   = Wvt + (size_t)Hc * DKc * Dc;

  // 0) precision/layout prep
  prep_x_kernel<<<(BNc * Dc / 4) / 256, 256, 0, stream>>>(q, h, qx, hx);
  prep_w_kernel<<<(4 * Hc * DKc * Dc) / 256, 256, 0, stream>>>(Wq, Wk, Wv, Wo,
                                                               Wqt, Wkt, Wvt, Wot);

  // 1) projections: 3*8*1024 = 24576 waves (16x64 stripe each), 8 waves/block
  proj_kernel<<<24576 / 8, 256, 0, stream>>>(qx, hx, Wqt, Wkt, Wvt, Qb, Kb, Vt);

  // 2) fused attention: H*B*(NQ/16) = 8192 blocks, 107,584 B dynamic LDS (< 320 KB/WG)
  attn_kernel<<<Hc * Bc * (NQc / 16), 256, ATTN_SMEM_BYTES, stream>>>(
      Qb, Kb, Vt, mask, attn_p, heads);

  // 3) output projection: 1024*8 = 8192 waves (16x64 stripe each)
  out_kernel<<<8192 / 8, 256, 0, stream>>>(heads, Wot, out_p);
}
